// NCNLinkPredictor_59803124630209
// MI455X (gfx1250) — compile-verified, matrix operand order checked by reading, not run on hardware
//
#include <hip/hip_runtime.h>
#include <hip/hip_bf16.h>

// ---------------------------------------------------------------------------
// NCN link predictor forward for MI455X (gfx1250).
// Dense GEMMs: v_wmma_f32_16x16x32_bf16 (bf16 in, f32 accumulate), A tiles
// streamed via GLOBAL_LOAD_ASYNC_TO_LDS_B128 (ASYNCcnt / s_wait_asynccnt)
// into double-buffered LDS; 32x64 wave tiles (2x4 WMMA) for high WMMA:LDS
// ratio; wave32 fragment layouts per the CDNA5 ISA.
// Graph attention / segment ops use f32 atomics + wave32 shuffle reductions.
// ---------------------------------------------------------------------------

typedef __attribute__((ext_vector_type(16))) __bf16 bf16x16;
typedef __attribute__((ext_vector_type(8)))  float  f32x8;

#define DEV static __device__ __forceinline__

constexpr int N_NODES = 20000;
constexpr int E_EDGES = 160000;
constexpr int P_PAIRS = 32768;
constexpr int IN_F    = 128;
constexpr int PROJ    = 256;
constexpr int HID     = 256;
constexpr int HEADS   = 4;
constexpr int C1      = 64;   // HID / HEADS
constexpr int DH      = 1024;
constexpr int E2      = E_EDGES + N_NODES;   // edges + self loops
constexpr int KD1     = 2 * HID + 3;         // 515
constexpr int KPAD    = 544;                 // 515 padded to multiple of 32
constexpr float EPS   = 1e-5f;
constexpr float NEG   = 0.2f;

// ------------------------------ small utils -------------------------------

__global__ __launch_bounds__(256) void k_zero_f32(float* p, size_t n) {
  size_t i = (size_t)blockIdx.x * blockDim.x + threadIdx.x;
  if (i < n) p[i] = 0.0f;
}

__global__ __launch_bounds__(256) void k_fill_u32(unsigned* p, unsigned v, size_t n) {
  size_t i = (size_t)blockIdx.x * blockDim.x + threadIdx.x;
  if (i < n) p[i] = v;
}

__global__ __launch_bounds__(256) void k_cvt_bf16(const float* s, __bf16* d, size_t n) {
  size_t i = (size_t)blockIdx.x * blockDim.x + threadIdx.x;
  if (i < n) d[i] = (__bf16)s[i];
}

// Pad+convert Wd1 (515x1024 f32) -> (544x1024 bf16), zero fill padded rows.
__global__ __launch_bounds__(256) void k_pad_cvt_wd1(const float* s, __bf16* d) {
  size_t i = (size_t)blockIdx.x * blockDim.x + threadIdx.x;
  size_t n = (size_t)KPAD * DH;
  if (i >= n) return;
  int r = (int)(i / DH), c = (int)(i % DH);
  d[i] = (r < KD1) ? (__bf16)s[(size_t)r * DH + c] : (__bf16)0.0f;
}

// ----------------------- CDNA5 async copy primitives ----------------------
// gfx1250 flat->LDS mapping truncates to addr[31:0], so the low 32 bits of a
// generic shared-memory pointer are the LDS byte address the instruction's
// VDST operand expects.
DEV void async_copy_b128(void* lds_dst, const void* gsrc) {
  unsigned loff = (unsigned)(uintptr_t)lds_dst;
  asm volatile("global_load_async_to_lds_b128 %0, %1, off"
               :
               : "v"(loff), "v"(gsrc)
               : "memory");
}
DEV void wait_asynccnt0() {
  asm volatile("s_wait_asynccnt 0x0" ::: "memory");
}

// ------------------------------ WMMA GEMM ---------------------------------
// C[M,N] = A[M,K] @ B[K,N] + bias  (A,B bf16 row-major, C f32)
// Block 256 threads = 8 waves as 4(m) x 2(n); wave tile 32x64 (2x4 WMMA).
// A tile: async global->LDS (double buffered).  B tile: staged transposed.
// Requires K % 32 == 0, N % 128 == 0; M ragged (guarded).

#define BM 128
#define BN 128
#define BK 32

__global__ __launch_bounds__(256) void k_gemm_bf16(
    const __bf16* __restrict__ A, const __bf16* __restrict__ B,
    const float* __restrict__ bias, float* __restrict__ C,
    int M, int N, int K) {
  __shared__ __bf16 sA[2][BM][BK];       // 2 x 8 KB
  __shared__ __bf16 sBt[2][BN][BK];      // 2 x 8 KB, B staged transposed

  const int tid  = threadIdx.x;
  const int wave = tid >> 5;
  const int lane = tid & 31;
  const int wm   = wave & 3;             // 0..3
  const int wn   = wave >> 2;            // 0..1
  const int bm0  = blockIdx.x * BM;
  const int bn0  = blockIdx.y * BN;

  f32x8 acc[2][4] = {};

  const int lrow  = lane & 15;
  const int khalf = (lane >= 16) ? 8 : 0;
  const int lhi   = (lane >= 16) ? 8 : 0;

  // B staging coords: 16 k-rows per thread at one column
  const int bc  = tid & 127;
  const int br0 = (tid >> 7) * 16;
  const int gc  = bn0 + bc;

  // Issue async copies for one 128x32 A tile into LDS buffer `buf`.
  auto issueA = [&](int k0, int buf) {
#pragma unroll
    for (int cc = 0; cc < 2; ++cc) {
      int chunk = tid + cc * 256;        // 512 chunks of 16B
      int r  = chunk >> 2;
      int c8 = (chunk & 3) * 8;
      int gr = bm0 + r;
      if (gr < M) {
        async_copy_b128(&sA[buf][r][c8], A + (size_t)gr * K + k0 + c8);
      } else {
        uint4 z{0u, 0u, 0u, 0u};
        *(uint4*)&sA[buf][r][c8] = z;
      }
    }
  };
  // Stage one 32x128 B tile transposed into sBt[buf].
  auto issueB = [&](int k0, int buf) {
    if (k0 + BK < K)
      __builtin_prefetch(B + (size_t)(k0 + BK + br0) * N + gc, 0, 3);
#pragma unroll
    for (int j = 0; j < 16; ++j)
      sBt[buf][bc][br0 + j] = B[(size_t)(k0 + br0 + j) * N + gc];
  };

  const int nIter = K / BK;
  issueA(0, 0);
  issueB(0, 0);

  for (int it = 0; it < nIter; ++it) {
    const int cur = it & 1;
    wait_asynccnt0();                    // tile `it` landed in LDS
    __syncthreads();                     // visible to all waves (covers B ds stores);
                                         // also: all waves done reading buffer 1-cur
    if (it + 1 < nIter) {                // prefetch next tile into other buffer
      issueA((it + 1) * BK, 1 - cur);
      issueB((it + 1) * BK, 1 - cur);
    }

#pragma unroll
    for (int tm = 0; tm < 2; ++tm) {
      union { bf16x16 v; __bf16 e[16]; } af;
      const int ar = wm * 32 + tm * 16 + lrow;
#pragma unroll
      for (int j = 0; j < 8; ++j) {
        af.e[j]     = sA[cur][ar][khalf + j];
        af.e[8 + j] = sA[cur][ar][16 + khalf + j];
      }
#pragma unroll
      for (int tn = 0; tn < 4; ++tn) {
        union { bf16x16 v; __bf16 e[16]; } bfr;
        const int bcol = wn * 64 + tn * 16 + lrow;
#pragma unroll
        for (int j = 0; j < 8; ++j) {
          bfr.e[j]     = sBt[cur][bcol][khalf + j];
          bfr.e[8 + j] = sBt[cur][bcol][16 + khalf + j];
        }
        acc[tm][tn] = __builtin_amdgcn_wmma_f32_16x16x32_bf16(
            false, af.v, false, bfr.v, (short)0, acc[tm][tn], false, false);
      }
    }
    // no trailing barrier: next iteration's wait+barrier orders buffer reuse
  }

  // --- store: C/D layout — VGPR r holds M = r + (lane>=16 ? 8 : 0), N = lane%16
#pragma unroll
  for (int tm = 0; tm < 2; ++tm) {
#pragma unroll
    for (int tn = 0; tn < 4; ++tn) {
      const int col = bn0 + wn * 64 + tn * 16 + lrow;
      const float bv = bias ? bias[col] : 0.0f;
#pragma unroll
      for (int r = 0; r < 8; ++r) {
        const int row = bm0 + wm * 32 + tm * 16 + r + lhi;
        if (row < M) C[(size_t)row * N + col] = acc[tm][tn][r] + bv;
      }
    }
  }
}

// --------------------------- LayerNorm + ELU ------------------------------
// One 256-thread block per row, C == 256.
__global__ __launch_bounds__(256) void k_ln_elu(float* H, const float* g,
                                                const float* b, int C) {
  const int row = blockIdx.x;
  const int t = threadIdx.x;
  float v = H[(size_t)row * C + t];
  __shared__ float red[256];
  red[t] = v; __syncthreads();
  for (int o = 128; o; o >>= 1) { if (t < o) red[t] += red[t + o]; __syncthreads(); }
  const float mu = red[0] / C; __syncthreads();
  const float d = v - mu;
  red[t] = d * d; __syncthreads();
  for (int o = 128; o; o >>= 1) { if (t < o) red[t] += red[t + o]; __syncthreads(); }
  const float var = red[0] / C;
  float x = d * rsqrtf(var + EPS) * g[t] + b[t];
  H[(size_t)row * C + t] = (x > 0.0f) ? x : (expf(x) - 1.0f);
}

// ------------------------- BatchNorm column stats -------------------------
__global__ __launch_bounds__(256) void k_colstats(const float* X, int M, int C,
                                                  float* mean, float* rstd) {
  const int col = blockIdx.x;
  float s = 0.0f, s2 = 0.0f;
  for (int r = threadIdx.x; r < M; r += blockDim.x) {
    float v = X[(size_t)r * C + col];
    s += v; s2 += v * v;
  }
  __shared__ float bs[256], bq[256];
  bs[threadIdx.x] = s; bq[threadIdx.x] = s2; __syncthreads();
  for (int o = 128; o; o >>= 1) {
    if (threadIdx.x < o) { bs[threadIdx.x] += bs[threadIdx.x + o]; bq[threadIdx.x] += bq[threadIdx.x + o]; }
    __syncthreads();
  }
  if (threadIdx.x == 0) {
    float mu = bs[0] / M;
    float var = bq[0] / M - mu * mu;
    mean[col] = mu;
    rstd[col] = rsqrtf(var + EPS);
  }
}

// BN apply + optional residual + ELU (all uses apply ELU), f32 out.
__global__ __launch_bounds__(256) void k_bn_elu(const float* X, const float* mean,
                                                const float* rstd, const float* g,
                                                const float* b, const float* res,
                                                float* out, size_t n, int C) {
  size_t i = (size_t)blockIdx.x * blockDim.x + threadIdx.x;
  if (i >= n) return;
  int col = (int)(i % C);
  float v = (X[i] - mean[col]) * rstd[col] * g[col] + b[col];
  if (res) v += res[i];
  out[i] = (v > 0.0f) ? v : (expf(v) - 1.0f);
}

// BN apply + ELU, bf16 out (feeds next WMMA GEMM directly).
__global__ __launch_bounds__(256) void k_bn_elu_bf16(const float* X, const float* mean,
                                                     const float* rstd, const float* g,
                                                     const float* b, __bf16* out,
                                                     size_t n, int C) {
  size_t i = (size_t)blockIdx.x * blockDim.x + threadIdx.x;
  if (i >= n) return;
  int col = (int)(i % C);
  float v = (X[i] - mean[col]) * rstd[col] * g[col] + b[col];
  v = (v > 0.0f) ? v : (expf(v) - 1.0f);
  out[i] = (__bf16)v;
}

// ------------------------------- GATv2 ops --------------------------------

__global__ __launch_bounds__(256) void k_build_edges(const int* ei, int* src2, int* dst2) {
  int i = blockIdx.x * blockDim.x + threadIdx.x;
  if (i < E_EDGES) { src2[i] = ei[i]; dst2[i] = ei[E_EDGES + i]; }
  else if (i < E2) { src2[i] = i - E_EDGES; dst2[i] = i - E_EDGES; }
}

__global__ __launch_bounds__(256) void k_add_bias(float* X, const float* bias,
                                                  size_t n, int C) {
  size_t i = (size_t)blockIdx.x * blockDim.x + threadIdx.x;
  if (i < n) X[i] += bias[i % C];
}

// logits[e,h] = sum_c att[h,c] * leaky_relu(xl[src,h,c] + xr[dst,h,c])
// One wave32 per (edge, head).
__global__ __launch_bounds__(256) void k_gat_logits(const float* xl, const float* xr,
                                                    const float* att, const int* src,
                                                    const int* dst, float* logits,
                                                    int c) {
  int gw = (blockIdx.x * blockDim.x + threadIdx.x) >> 5;
  int lane = threadIdx.x & 31;
  int e = gw / HEADS, h = gw % HEADS;
  if (e >= E2) return;
  const int s = src[e], d = dst[e];
  const float* pl = xl + (size_t)s * HEADS * c + (size_t)h * c;
  const float* pr = xr + (size_t)d * HEADS * c + (size_t)h * c;
  const float* pa = att + (size_t)h * c;
  float acc = 0.0f;
  for (int i = lane; i < c; i += 32) {
    float v = pl[i] + pr[i];
    v = (v > 0.0f) ? v : v * NEG;
    acc += v * pa[i];
  }
  for (int o = 16; o; o >>= 1) acc += __shfl_down(acc, o, 32);
  if (lane == 0) logits[(size_t)e * HEADS + h] = acc;
}

DEV unsigned flip_f32(float f) {
  unsigned u = __float_as_uint(f);
  return (u & 0x80000000u) ? ~u : (u | 0x80000000u);
}
DEV float unflip_f32(unsigned u) {
  u = (u & 0x80000000u) ? (u & 0x7FFFFFFFu) : ~u;
  return __uint_as_float(u);
}

__global__ __launch_bounds__(256) void k_seg_max(const float* logits, const int* dst,
                                                 unsigned* mmax) {
  int i = blockIdx.x * blockDim.x + threadIdx.x;   // (e,h)
  if (i >= E2 * HEADS) return;
  int e = i / HEADS, h = i % HEADS;
  atomicMax(&mmax[(size_t)dst[e] * HEADS + h], flip_f32(logits[i]));
}

__global__ __launch_bounds__(256) void k_softmax_p(const float* logits, const unsigned* mmax,
                                                   const int* dst, float* p, float* denom) {
  int i = blockIdx.x * blockDim.x + threadIdx.x;   // (e,h)
  if (i >= E2 * HEADS) return;
  int e = i / HEADS, h = i % HEADS;
  float m = unflip_f32(mmax[(size_t)dst[e] * HEADS + h]);
  float pe = expf(logits[i] - m);
  p[i] = pe;
  atomicAdd(&denom[(size_t)dst[e] * HEADS + h], pe);
}

// out[dst, h, c] += xl[src, h, c] * alpha[e, h]
__global__ __launch_bounds__(256) void k_scatter(const float* xl, const float* p,
                                                 const float* denom, const int* src,
                                                 const int* dst, float* out, int c) {
  size_t i = (size_t)blockIdx.x * blockDim.x + threadIdx.x;
  size_t total = (size_t)E2 * HEADS * c;
  if (i >= total) return;
  int hc = HEADS * c;
  int e = (int)(i / hc), rem = (int)(i % hc), h = rem / c;
  int s = src[e], d = dst[e];
  float alpha = p[(size_t)e * HEADS + h] /
                (denom[(size_t)d * HEADS + h] + 1e-16f);
  atomicAdd(&out[(size_t)d * hc + rem], xl[(size_t)s * hc + rem] * alpha);
}

// Layer-2: mean over heads + bias -> [N, 256]
__global__ __launch_bounds__(256) void k_headmean(const float* agg, const float* cb,
                                                  float* z0) {
  size_t i = (size_t)blockIdx.x * blockDim.x + threadIdx.x;
  if (i >= (size_t)N_NODES * HID) return;
  int n = (int)(i / HID), c = (int)(i % HID);
  float s = 0.0f;
#pragma unroll
  for (int h = 0; h < HEADS; ++h) s += agg[(size_t)n * HEADS * HID + (size_t)h * HID + c];
  z0[i] = s * (1.0f / HEADS) + cb[c];
}

// ------------------------------ decoder side ------------------------------

__global__ __launch_bounds__(256) void k_struct_proj(const float* raw, const float* Ws,
                                                     const float* bs, float* s0) {
  int pidx = blockIdx.x * blockDim.x + threadIdx.x;
  if (pidx >= P_PAIRS) return;
  float r0 = raw[pidx * 3 + 0], r1 = raw[pidx * 3 + 1], r2 = raw[pidx * 3 + 2];
#pragma unroll
  for (int j = 0; j < 3; ++j)
    s0[pidx * 3 + j] = r0 * Ws[0 * 3 + j] + r1 * Ws[1 * 3 + j] + r2 * Ws[2 * 3 + j] + bs[j];
}

// Hcat[p, 0:256]=z1, [256:512]=z2, [512:515]=struct, [515:544]=0  (bf16)
__global__ __launch_bounds__(256) void k_build_pairs(const float* z, const float* structf,
                                                     const int* pairs, __bf16* Hcat) {
  size_t i = (size_t)blockIdx.x * blockDim.x + threadIdx.x;
  if (i >= (size_t)P_PAIRS * KPAD) return;
  int pidx = (int)(i / KPAD), c = (int)(i % KPAD);
  int u = pairs[pidx * 2 + 0], v = pairs[pidx * 2 + 1];
  int a = (u > v) ? v : u;
  int b = (u > v) ? u : v;
  float val;
  if      (c < 256) val = z[(size_t)a * HID + c];
  else if (c < 512) val = z[(size_t)b * HID + (c - 256)];
  else if (c < 515) val = structf[pidx * 3 + (c - 512)];
  else              val = 0.0f;
  Hcat[i] = (__bf16)val;
}

// out[p] = d2[p,:] . Wd3 + bd3 — one wave32 per row.
__global__ __launch_bounds__(256) void k_final_dot(const float* d2, const float* Wd3,
                                                   const float* bd3, float* out) {
  int gw = (blockIdx.x * blockDim.x + threadIdx.x) >> 5;
  int lane = threadIdx.x & 31;
  if (gw >= P_PAIRS) return;
  const float* row = d2 + (size_t)gw * DH;
  float acc = 0.0f;
  for (int i = lane; i < DH; i += 32) acc += row[i] * Wd3[i];
  for (int o = 16; o; o >>= 1) acc += __shfl_down(acc, o, 32);
  if (lane == 0) out[gw] = acc + bd3[0];
}

// ------------------------------ host driver -------------------------------

static inline void* wsal(char*& p, size_t bytes) {
  void* r = (void*)p;
  p += (bytes + 255) & ~size_t(255);
  return r;
}
static inline size_t cdiv(size_t a, size_t b) { return (a + b - 1) / b; }

extern "C" void kernel_launch(void* const* d_in, const int* in_sizes, int n_in,
                              void* d_out, int out_size, void* d_ws, size_t ws_size,
                              hipStream_t stream) {
  // --- inputs (setup_inputs dict order, params flattened in insertion order)
  const float* x          = (const float*)d_in[0];   // 20000x128
  const float* struct_raw = (const float*)d_in[1];   // 32768x3
  const float* Wp    = (const float*)d_in[2];
  const float* bp    = (const float*)d_in[3];
  const float* ln_g  = (const float*)d_in[4];
  const float* ln_b  = (const float*)d_in[5];
  const float* Wl1   = (const float*)d_in[6];
  const float* bl1   = (const float*)d_in[7];
  const float* Wr1   = (const float*)d_in[8];
  const float* br1   = (const float*)d_in[9];
  const float* att1  = (const float*)d_in[10];
  const float* cb1   = (const float*)d_in[11];
  const float* bn1_g = (const float*)d_in[12];
  const float* bn1_b = (const float*)d_in[13];
  const float* Wl2   = (const float*)d_in[14];
  const float* bl2   = (const float*)d_in[15];
  const float* Wr2   = (const float*)d_in[16];
  const float* br2   = (const float*)d_in[17];
  const float* att2  = (const float*)d_in[18];
  const float* cb2   = (const float*)d_in[19];
  const float* bn2_g = (const float*)d_in[20];
  const float* bn2_b = (const float*)d_in[21];
  const float* Ws    = (const float*)d_in[22];
  const float* bs_   = (const float*)d_in[23];
  const float* bns_g = (const float*)d_in[24];
  const float* bns_b = (const float*)d_in[25];
  const float* Wd1   = (const float*)d_in[26];
  const float* bd1   = (const float*)d_in[27];
  const float* bnd1_g= (const float*)d_in[28];
  const float* bnd1_b= (const float*)d_in[29];
  const float* Wd2   = (const float*)d_in[30];
  const float* bd2   = (const float*)d_in[31];
  const float* bnd2_g= (const float*)d_in[32];
  const float* bnd2_b= (const float*)d_in[33];
  const float* Wd3   = (const float*)d_in[34];
  const float* bd3   = (const float*)d_in[35];
  const int* edge_index = (const int*)d_in[36];      // 2x160000
  const int* edge_pairs = (const int*)d_in[37];      // 32768x2
  float* out = (float*)d_out;

  // --- workspace carve (bump allocator; deterministic)
  char* wp = (char*)d_ws;
  __bf16* xbf    = (__bf16*)wsal(wp, (size_t)N_NODES * IN_F * 2);
  __bf16* Wp_b   = (__bf16*)wsal(wp, (size_t)IN_F * PROJ * 2);
  __bf16* Wl1_b  = (__bf16*)wsal(wp, (size_t)PROJ * HID * 2);
  __bf16* Wr1_b  = (__bf16*)wsal(wp, (size_t)PROJ * HID * 2);
  __bf16* Wl2_b  = (__bf16*)wsal(wp, (size_t)HID * (HEADS * HID) * 2);
  __bf16* Wr2_b  = (__bf16*)wsal(wp, (size_t)HID * (HEADS * HID) * 2);
  __bf16* Wd1_b  = (__bf16*)wsal(wp, (size_t)KPAD * DH * 2);
  __bf16* Wd2_b  = (__bf16*)wsal(wp, (size_t)DH * DH * 2);
  float*  hA     = (float*)wsal(wp, (size_t)N_NODES * HID * 4);     // h0 raw / z0 / z
  float*  hB     = (float*)wsal(wp, (size_t)N_NODES * HID * 4);     // h1 (residual)
  __bf16* hbf    = (__bf16*)wsal(wp, (size_t)N_NODES * HID * 2);
  float*  xl     = (float*)wsal(wp, (size_t)N_NODES * HEADS * HID * 4);
  float*  xr     = (float*)wsal(wp, (size_t)N_NODES * HEADS * HID * 4);
  float*  agg    = (float*)wsal(wp, (size_t)N_NODES * HEADS * HID * 4);
  int*    src2   = (int*)wsal(wp, (size_t)E2 * 4);
  int*    dst2   = (int*)wsal(wp, (size_t)E2 * 4);
  float*  logits = (float*)wsal(wp, (size_t)E2 * HEADS * 4);
  float*  pbuf   = (float*)wsal(wp, (size_t)E2 * HEADS * 4);
  unsigned* mmax = (unsigned*)wsal(wp, (size_t)N_NODES * HEADS * 4);
  float*  denom  = (float*)wsal(wp, (size_t)N_NODES * HEADS * 4);
  float*  mean   = (float*)wsal(wp, (size_t)DH * 4);
  float*  rstd   = (float*)wsal(wp, (size_t)DH * 4);
  float*  s0     = (float*)wsal(wp, (size_t)P_PAIRS * 3 * 4);
  __bf16* Hcat   = (__bf16*)wsal(wp, (size_t)P_PAIRS * KPAD * 2);
  float*  dbig   = (float*)wsal(wp, (size_t)P_PAIRS * DH * 4);      // d1raw, then d2raw/d2
  __bf16* dbf    = (__bf16*)wsal(wp, (size_t)P_PAIRS * DH * 2);
  (void)ws_size; (void)n_in; (void)in_sizes; (void)out_size;

  auto g1 = [](size_t n) { return dim3((unsigned)cdiv(n, 256)); };

  // --- 0) weight / input conversion to bf16
  k_cvt_bf16<<<g1((size_t)N_NODES * IN_F), 256, 0, stream>>>(x, xbf, (size_t)N_NODES * IN_F);
  k_cvt_bf16<<<g1((size_t)IN_F * PROJ), 256, 0, stream>>>(Wp, Wp_b, (size_t)IN_F * PROJ);
  k_cvt_bf16<<<g1((size_t)PROJ * HID), 256, 0, stream>>>(Wl1, Wl1_b, (size_t)PROJ * HID);
  k_cvt_bf16<<<g1((size_t)PROJ * HID), 256, 0, stream>>>(Wr1, Wr1_b, (size_t)PROJ * HID);
  k_cvt_bf16<<<g1((size_t)HID * HEADS * HID), 256, 0, stream>>>(Wl2, Wl2_b, (size_t)HID * HEADS * HID);
  k_cvt_bf16<<<g1((size_t)HID * HEADS * HID), 256, 0, stream>>>(Wr2, Wr2_b, (size_t)HID * HEADS * HID);
  k_pad_cvt_wd1<<<g1((size_t)KPAD * DH), 256, 0, stream>>>(Wd1, Wd1_b);
  k_cvt_bf16<<<g1((size_t)DH * DH), 256, 0, stream>>>(Wd2, Wd2_b, (size_t)DH * DH);

  // --- 1) projection + LN + ELU
  {
    dim3 grid((unsigned)cdiv(N_NODES, BM), PROJ / BN);
    k_gemm_bf16<<<grid, 256, 0, stream>>>(xbf, Wp_b, bp, hA, N_NODES, PROJ, IN_F);
  }
  k_ln_elu<<<N_NODES, 256, 0, stream>>>(hA, ln_g, ln_b, PROJ);
  k_cvt_bf16<<<g1((size_t)N_NODES * PROJ), 256, 0, stream>>>(hA, hbf, (size_t)N_NODES * PROJ);

  // --- 2) GAT layer 1 (heads=4, c=64, concat)
  {
    dim3 grid((unsigned)cdiv(N_NODES, BM), HID / BN);
    k_gemm_bf16<<<grid, 256, 0, stream>>>(hbf, Wl1_b, bl1, xl, N_NODES, HID, PROJ);
    k_gemm_bf16<<<grid, 256, 0, stream>>>(hbf, Wr1_b, br1, xr, N_NODES, HID, PROJ);
  }
  k_build_edges<<<g1(E2), 256, 0, stream>>>(edge_index, src2, dst2);
  k_fill_u32<<<g1((size_t)N_NODES * HEADS), 256, 0, stream>>>(mmax, 0u, (size_t)N_NODES * HEADS);
  k_zero_f32<<<g1((size_t)N_NODES * HEADS), 256, 0, stream>>>(denom, (size_t)N_NODES * HEADS);
  k_zero_f32<<<g1((size_t)N_NODES * HID), 256, 0, stream>>>(agg, (size_t)N_NODES * HID);
  k_gat_logits<<<g1((size_t)E2 * HEADS * 32), 256, 0, stream>>>(xl, xr, att1, src2, dst2, logits, C1);
  k_seg_max<<<g1((size_t)E2 * HEADS), 256, 0, stream>>>(logits, dst2, mmax);
  k_softmax_p<<<g1((size_t)E2 * HEADS), 256, 0, stream>>>(logits, mmax, dst2, pbuf, denom);
  k_scatter<<<g1((size_t)E2 * HEADS * C1), 256, 0, stream>>>(xl, pbuf, denom, src2, dst2, agg, C1);
  k_add_bias<<<g1((size_t)N_NODES * HID), 256, 0, stream>>>(agg, cb1, (size_t)N_NODES * HID, HID);
  k_colstats<<<HID, 256, 0, stream>>>(agg, N_NODES, HID, mean, rstd);
  k_bn_elu<<<g1((size_t)N_NODES * HID), 256, 0, stream>>>(agg, mean, rstd, bn1_g, bn1_b,
                                                          nullptr, hB, (size_t)N_NODES * HID, HID);
  k_cvt_bf16<<<g1((size_t)N_NODES * HID), 256, 0, stream>>>(hB, hbf, (size_t)N_NODES * HID);

  // --- 3) GAT layer 2 (heads=4, c=256, mean) with residual
  {
    dim3 grid((unsigned)cdiv(N_NODES, BM), (HEADS * HID) / BN);
    k_gemm_bf16<<<grid, 256, 0, stream>>>(hbf, Wl2_b, bl2, xl, N_NODES, HEADS * HID, HID);
    k_gemm_bf16<<<grid, 256, 0, stream>>>(hbf, Wr2_b, br2, xr, N_NODES, HEADS * HID, HID);
  }
  k_fill_u32<<<g1((size_t)N_NODES * HEADS), 256, 0, stream>>>(mmax, 0u, (size_t)N_NODES * HEADS);
  k_zero_f32<<<g1((size_t)N_NODES * HEADS), 256, 0, stream>>>(denom, (size_t)N_NODES * HEADS);
  k_zero_f32<<<g1((size_t)N_NODES * HEADS * HID), 256, 0, stream>>>(agg, (size_t)N_NODES * HEADS * HID);
  k_gat_logits<<<g1((size_t)E2 * HEADS * 32), 256, 0, stream>>>(xl, xr, att2, src2, dst2, logits, HID);
  k_seg_max<<<g1((size_t)E2 * HEADS), 256, 0, stream>>>(logits, dst2, mmax);
  k_softmax_p<<<g1((size_t)E2 * HEADS), 256, 0, stream>>>(logits, mmax, dst2, pbuf, denom);
  k_scatter<<<g1((size_t)E2 * HEADS * HID), 256, 0, stream>>>(xl, pbuf, denom, src2, dst2, agg, HID);
  k_headmean<<<g1((size_t)N_NODES * HID), 256, 0, stream>>>(agg, cb2, hA);
  k_colstats<<<HID, 256, 0, stream>>>(hA, N_NODES, HID, mean, rstd);
  k_bn_elu<<<g1((size_t)N_NODES * HID), 256, 0, stream>>>(hA, mean, rstd, bn2_g, bn2_b,
                                                          hB, hA, (size_t)N_NODES * HID, HID);  // z

  // --- 4) struct branch
  k_struct_proj<<<g1(P_PAIRS), 256, 0, stream>>>(struct_raw, Ws, bs_, s0);
  k_colstats<<<3, 256, 0, stream>>>(s0, P_PAIRS, 3, mean, rstd);
  k_bn_elu<<<g1((size_t)P_PAIRS * 3), 256, 0, stream>>>(s0, mean, rstd, bns_g, bns_b,
                                                        nullptr, s0, (size_t)P_PAIRS * 3, 3);

  // --- 5) pair features + decoder MLP
  k_build_pairs<<<g1((size_t)P_PAIRS * KPAD), 256, 0, stream>>>(hA, s0, edge_pairs, Hcat);
  {
    dim3 grid((unsigned)cdiv(P_PAIRS, BM), DH / BN);
    k_gemm_bf16<<<grid, 256, 0, stream>>>(Hcat, Wd1_b, bd1, dbig, P_PAIRS, DH, KPAD);
  }
  k_colstats<<<DH, 256, 0, stream>>>(dbig, P_PAIRS, DH, mean, rstd);
  k_bn_elu_bf16<<<g1((size_t)P_PAIRS * DH), 256, 0, stream>>>(dbig, mean, rstd, bnd1_g, bnd1_b,
                                                              dbf, (size_t)P_PAIRS * DH, DH);
  {
    dim3 grid((unsigned)cdiv(P_PAIRS, BM), DH / BN);
    k_gemm_bf16<<<grid, 256, 0, stream>>>(dbf, Wd2_b, bd2, dbig, P_PAIRS, DH, DH);
  }
  k_colstats<<<DH, 256, 0, stream>>>(dbig, P_PAIRS, DH, mean, rstd);
  k_bn_elu<<<g1((size_t)P_PAIRS * DH), 256, 0, stream>>>(dbig, mean, rstd, bnd2_g, bnd2_b,
                                                         nullptr, dbig, (size_t)P_PAIRS * DH, DH);
  k_final_dot<<<g1((size_t)P_PAIRS * 32), 256, 0, stream>>>(dbig, Wd3, bd3, out);
}
